// LSTMTagger_76699525972249
// MI455X (gfx1250) — matrix operand order repaired and checked by reading, not compiled, not run on hardware
//
#include <hip/hip_runtime.h>
#include <hip/hip_bf16.h>
#include <math.h>

// ---------------------------------------------------------------------------
// LSTM tagger, CDNA5 / gfx1250 (wave32, WMMA 16x16x32 bf16, f32 accumulate)
//   B=32 T=512 V=32000 E=512 H=1024 O=128
// GEMM operands pre-swizzled into the exact WMMA VGPR fragment layout (bf16):
// steady-state inner loops are 2xb128(A) + 2xb128(B) + v_wmma, dual
// accumulators to overlap back-to-back WMMAs. Embedding gather staged through
// LDS once per block (8x traffic cut). Persistent cooperative LSTM with
// per-step device-scope split barrier; W_hh slice + h ping-pong L2 resident.
// ---------------------------------------------------------------------------

typedef __attribute__((ext_vector_type(16))) __bf16 v16bf;
typedef __attribute__((ext_vector_type(2)))  __bf16 v2bf;
typedef __attribute__((ext_vector_type(8)))  float  v8f;

#define B_   32
#define T_   512
#define E_   512
#define H_   1024
#define O_   128
#define G4H  4096
#define NWG_LSTM 64

// packed tile geometry: one 16(row/col) x 32(k) fragment tile = [lane 32][e 16]
#define TILE_ELEMS 512            // 32 lanes * 16 halves of bf16

static __device__ __forceinline__ float sigmoidf_(float x) {
  return 1.0f / (1.0f + __expf(-x));
}

// inverse of the A-fragment k striping (ISA 7.12.2, 16-bit A 16x32):
//   half 0 covers k in {0..7, 16..23}, half 1 covers {8..15, 24..31}
static __device__ __forceinline__ void a_coord(int kp, int& half, int& e) {
  half = (kp >> 3) & 1;
  const int base = kp - 8 * half;
  const int v = (base >= 16) ? (4 + ((base - 16) >> 1)) : (base >> 1);
  e = 2 * v + (kp & 1);
}

// ====================== one-time operand packing ===========================
// W row-major [N x ldw] fp32 -> B-fragment layout [nt][kc][lane][16] bf16
// B inverse mapping: k = c*32 + 8*(lane>>4) + (e<8 ? e : e+8) ; n = nt*16 + (lane&15)
__global__ __launch_bounds__(256) void pack_b_kernel(
    const float* __restrict__ W, __bf16* __restrict__ out, int KC, int ldw,
    int koff) {
  const size_t gid = (size_t)blockIdx.x * 256 + threadIdx.x;
  const int e    = gid & 15;
  const int lane = (gid >> 4) & 31;
  const int cc   = (int)((gid >> 9) % KC);
  const int nt   = (int)((gid >> 9) / KC);
  const int n = nt * 16 + (lane & 15);
  const int k = cc * 32 + ((lane >> 4) << 3) + (e < 8 ? e : e + 8);
  out[gid] = (__bf16)W[(size_t)n * ldw + koff + k];
}

// ============================ Phase 1 ======================================
// x_proj = gather(emb)[B*T,E] @ W_ih^T + bias
// Output layout: xproj2[t][wg][gate][b*16+jj]  (one 4KB block per WG per step)
// grid (B*T/16, 4H/128), block 256 (8 waves). Gathered A tile staged in LDS.
__global__ __launch_bounds__(256) void embed_xproj_kernel(
    const int* __restrict__ tokens, const float* __restrict__ emb,
    const __bf16* __restrict__ wih_pack, const float* __restrict__ b_ih,
    const float* __restrict__ b_hh, __bf16* __restrict__ xproj2) {
  __shared__ __bf16 atile[16][E_];    // 16 gathered rows, bf16 (16 KB)

  const int tid  = threadIdx.x;
  const int wave = tid >> 5;
  const int lane = tid & 31;
  const int half = lane >> 4;
  const int l16  = lane & 15;
  const int row0 = blockIdx.x * 16;                 // rows = b*T + t
  const int nt   = blockIdx.y * 8 + wave;           // 16-col tile of 4H
  const int n0   = nt * 16;

  // stage gathered rows fp32 -> bf16 into LDS (each thread: 32 elements)
  {
    const int r  = tid >> 4;
    const int c0 = (tid & 15) * 32;
    const float* src = emb + (size_t)tokens[row0 + r] * E_ + c0;
    v16bf t0, t1;
#pragma unroll
    for (int i = 0; i < 16; ++i) {
      t0[i] = (__bf16)src[i];
      t1[i] = (__bf16)src[16 + i];
    }
    *(v16bf*)&atile[r][c0]      = t0;
    *(v16bf*)&atile[r][c0 + 16] = t1;
  }
  __syncthreads();

  const __bf16* aptr = &atile[l16][8 * half];
  const __bf16* bptr = wih_pack + (size_t)nt * (E_ / 32) * TILE_ELEMS +
                       lane * 16;

  v8f acc0 = {}, acc1 = {};
#pragma unroll
  for (int c = 0; c < E_ / 32; c += 2) {
    v16bf a0, a1;
#pragma unroll
    for (int v = 0; v < 8; ++v) {
      const int ka = ((v < 4) ? 2 * v : 2 * v + 8);
      const v2bf e0 = *(const v2bf*)(aptr + c * 32 + ka);
      const v2bf e1 = *(const v2bf*)(aptr + c * 32 + 32 + ka);
      a0[2 * v] = e0.x; a0[2 * v + 1] = e0.y;
      a1[2 * v] = e1.x; a1[2 * v + 1] = e1.y;
    }
    const v16bf b0 = *(const v16bf*)(bptr + c * TILE_ELEMS);
    const v16bf b1 = *(const v16bf*)(bptr + (c + 1) * TILE_ELEMS);
    acc0 = __builtin_amdgcn_wmma_f32_16x16x32_bf16(false, a0, false, b0,
                                                   (short)0, acc0, false, false);
    acc1 = __builtin_amdgcn_wmma_f32_16x16x32_bf16(false, a1, false, b1,
                                                   (short)0, acc1, false, false);
  }
  const float bias = b_ih[n0 + l16] + b_hh[n0 + l16];
  const int gate = n0 >> 10;              // which gate block of 4H
  const int wg   = (n0 & (H_ - 1)) >> 4;  // which hidden 16-slice
#pragma unroll
  for (int r = 0; r < 8; ++r) {
    const int row = row0 + r + 8 * half;   // C layout: VGPR r -> M = r + 8*half
    const int bb  = row >> 9;              // row / T  (batch)
    const int tt  = row & (T_ - 1);        // row % T
    xproj2[((((size_t)tt * NWG_LSTM + wg) * 4 + gate) * 32 + bb) * 16 + l16] =
        (__bf16)(acc0[r] + acc1[r] + bias);
  }
}

// ============================ Phase 2 ======================================
// Persistent cooperative kernel: 64 WGs, WG w owns hidden slice j0 = 16*w and
// computes all 4 gates for it. 8 waves = (m-tile 0/1) x gate. h is kept
// double-buffered in A-fragment layout (hpack, 128 KB -> L2 resident).
__global__ __launch_bounds__(256, 1) void lstm_persistent_kernel(
    const __bf16* __restrict__ whh_pack, const __bf16* __restrict__ xproj2,
    __bf16* __restrict__ hpack, __bf16* __restrict__ hs_pack,
    unsigned int* __restrict__ bar) {
  __shared__ float gsh[4][32][16];   // gate pre-activations [gate][batch][jj]
  __shared__ float csh[32][16];      // persistent cell state

  const int tid  = threadIdx.x;
  const int wave = tid >> 5;
  const int lane = tid & 31;
  const int half = lane >> 4;
  const int l16  = lane & 15;
  const int mt   = wave & 1;
  const int gate = wave >> 1;
  const int j0   = blockIdx.x * 16;

  // A fragments: hpack[par][mt][chunk][lane][16]
  const __bf16* apack0 = hpack + (size_t)mt * (32 * TILE_ELEMS) + lane * 16;
  // B fragments: whh_pack[nt = gate*64 + wg][chunk][lane][16]
  const __bf16* bpack = whh_pack +
                        (size_t)(gate * 64 + blockIdx.x) * (32 * TILE_ELEMS) +
                        lane * 16;

  // zero cell state and both parities of hpack (cooperatively, 2KB/WG)
  for (int e = tid; e < 512; e += 256) csh[e >> 4][e & 15] = 0.0f;
  {
    __bf16* z = hpack + (size_t)blockIdx.x * 1024;
    for (int i = tid; i < 1024; i += 256) z[i] = (__bf16)0.0f;
  }
  __threadfence();
  __syncthreads();
  if (tid == 0) {
    __hip_atomic_fetch_add(bar, 1u, __ATOMIC_RELEASE, __HIP_MEMORY_SCOPE_AGENT);
    while (__hip_atomic_load(bar, __ATOMIC_ACQUIRE, __HIP_MEMORY_SCOPE_AGENT) <
           (unsigned)NWG_LSTM)
      __builtin_amdgcn_s_sleep(1);
  }
  __syncthreads();

  for (int t = 0; t < T_; ++t) {
    const int par = t & 1;
    const __bf16* apack = apack0 + (size_t)par * (2 * 32 * TILE_ELEMS);

    v8f acc0 = {}, acc1 = {};
#pragma unroll 8
    for (int c = 0; c < 32; c += 2) {
      const v16bf a0 = *(const v16bf*)(apack + c * TILE_ELEMS);
      const v16bf b0 = *(const v16bf*)(bpack + c * TILE_ELEMS);
      const v16bf a1 = *(const v16bf*)(apack + (c + 1) * TILE_ELEMS);
      const v16bf b1 = *(const v16bf*)(bpack + (c + 1) * TILE_ELEMS);
      acc0 = __builtin_amdgcn_wmma_f32_16x16x32_bf16(false, a0, false, b0,
                                                     (short)0, acc0, false,
                                                     false);
      acc1 = __builtin_amdgcn_wmma_f32_16x16x32_bf16(false, a1, false, b1,
                                                     (short)0, acc1, false,
                                                     false);
    }
#pragma unroll
    for (int r = 0; r < 8; ++r)
      gsh[gate][mt * 16 + r + 8 * half][l16] = acc0[r] + acc1[r];
    __syncthreads();

    // fused gate nonlinearities + cell/hidden update; write h pre-swizzled
    const size_t nextbase = (size_t)(par ^ 1) * (2 * 32 * TILE_ELEMS);
    const __bf16* xp = xproj2 +
                       ((size_t)t * NWG_LSTM + blockIdx.x) * 4 * 512;
    for (int e = tid; e < 512; e += 256) {
      const int m = e >> 4, jj = e & 15;
      const int j = j0 + jj;
      const float gi = gsh[0][m][jj] + (float)xp[e];
      const float gf = gsh[1][m][jj] + (float)xp[e + 512];
      const float gg = gsh[2][m][jj] + (float)xp[e + 1024];
      const float go = gsh[3][m][jj] + (float)xp[e + 1536];
      const float i = sigmoidf_(gi);
      const float f = sigmoidf_(gf);
      const float g = tanhf(gg);
      const float o = sigmoidf_(go);
      const float c = f * csh[m][jj] + i * g;
      csh[m][jj] = c;
      const float h = o * tanhf(c);
      const __bf16 hb = (__bf16)h;

      int ah, ae;
      a_coord(j & 31, ah, ae);
      const int cc = j >> 5;
      // next-step recurrence operand (A layout over batch rows)
      hpack[nextbase + (size_t)(m >> 4) * (32 * TILE_ELEMS) + cc * TILE_ELEMS +
            ((m & 15) + 16 * ah) * 16 + ae] = hb;
      // logits operand (A layout over rows b*T + t)
      const int row = m * T_ + t;
      hs_pack[(size_t)(row >> 4) * (32 * TILE_ELEMS) + cc * TILE_ELEMS +
              ((row & 15) + 16 * ah) * 16 + ae] = hb;
    }
    // device-scope split barrier: monotonic counter, target NWG*(t+2)
    __threadfence();
    __syncthreads();
    if (tid == 0) {
      __hip_atomic_fetch_add(bar, 1u, __ATOMIC_RELEASE, __HIP_MEMORY_SCOPE_AGENT);
      const unsigned target = (unsigned)NWG_LSTM * (unsigned)(t + 2);
      while (__hip_atomic_load(bar, __ATOMIC_ACQUIRE,
                               __HIP_MEMORY_SCOPE_AGENT) < target)
        __builtin_amdgcn_s_sleep(1);
    }
    __syncthreads();
  }
}

// ============================ Phase 3a =====================================
// vproj[b,o] = b_lin[o] + hs[b, verb_idx[b]] . W_lin[o, 0:H]   (tiny)
__global__ __launch_bounds__(256) void vproj_kernel(
    const int* __restrict__ vidx, const __bf16* __restrict__ hs_pack,
    const float* __restrict__ W_lin, const float* __restrict__ b_lin,
    float* __restrict__ vproj) {
  const int gid = blockIdx.x * 256 + threadIdx.x;   // 4096 = B*O threads
  const int b = gid >> 7;
  const int o = gid & (O_ - 1);
  const int row = b * T_ + vidx[b];
  const int rb = row >> 4, mrow = row & 15;
  const __bf16* hrow = hs_pack + (size_t)rb * (32 * TILE_ELEMS);
  const float* wrow = W_lin + (size_t)o * (2 * H_);
  float acc = b_lin[o];
  for (int j = 0; j < H_; ++j) {
    int ah, ae;
    a_coord(j & 31, ah, ae);
    const float hv =
        (float)hrow[(j >> 5) * TILE_ELEMS + (mrow + 16 * ah) * 16 + ae];
    acc += hv * wrow[j];
  }
  vproj[gid] = acc;
}

// ============================ Phase 3b =====================================
// logits GEMM [16-row tile x 128 cols] via 8 waves, fused log-softmax (O=128)
__global__ __launch_bounds__(256) void logits_kernel(
    const __bf16* __restrict__ hs_pack, const __bf16* __restrict__ wo_pack,
    const float* __restrict__ vproj, float* __restrict__ out) {
  __shared__ float lsh[16][O_];
  const int tid  = threadIdx.x;
  const int wave = tid >> 5;
  const int lane = tid & 31;
  const int half = lane >> 4;
  const int l16  = lane & 15;
  const int row0 = blockIdx.x * 16;     // rows = b*T + t  (16 rows share b)
  const int o0   = wave * 16;

  const __bf16* apack =
      hs_pack + (size_t)blockIdx.x * (32 * TILE_ELEMS) + lane * 16;
  const __bf16* bpack =
      wo_pack + (size_t)wave * (32 * TILE_ELEMS) + lane * 16;

  v8f acc0 = {}, acc1 = {};
#pragma unroll 8
  for (int c = 0; c < 32; c += 2) {
    __builtin_prefetch(apack + (c + 4) * TILE_ELEMS, 0, 3);
    const v16bf a0 = *(const v16bf*)(apack + c * TILE_ELEMS);
    const v16bf b0 = *(const v16bf*)(bpack + c * TILE_ELEMS);
    const v16bf a1 = *(const v16bf*)(apack + (c + 1) * TILE_ELEMS);
    const v16bf b1 = *(const v16bf*)(bpack + (c + 1) * TILE_ELEMS);
    acc0 = __builtin_amdgcn_wmma_f32_16x16x32_bf16(false, a0, false, b0,
                                                   (short)0, acc0, false, false);
    acc1 = __builtin_amdgcn_wmma_f32_16x16x32_bf16(false, a1, false, b1,
                                                   (short)0, acc1, false, false);
  }
  const int bb = row0 >> 9;              // whole tile shares batch index
  const float vp = vproj[bb * O_ + o0 + l16];
#pragma unroll
  for (int r = 0; r < 8; ++r)
    lsh[r + 8 * half][o0 + l16] = acc0[r] + acc1[r] + vp;
  __syncthreads();
  if (tid < 16) {
    const int row = row0 + tid;
    float mx = -1e30f;
    for (int o = 0; o < O_; ++o) mx = fmaxf(mx, lsh[tid][o]);
    float s = 0.0f;
    for (int o = 0; o < O_; ++o) s += __expf(lsh[tid][o] - mx);
    const float lse = mx + __logf(s);
    float* orow = out + (size_t)row * O_;
    for (int o = 0; o < O_; ++o) orow[o] = lsh[tid][o] - lse;
  }
}

// ============================ Launch =======================================
extern "C" void kernel_launch(void* const* d_in, const int* in_sizes, int n_in,
                              void* d_out, int out_size, void* d_ws,
                              size_t ws_size, hipStream_t stream) {
  const int*   tokens = (const int*)d_in[0];
  const int*   vidx   = (const int*)d_in[1];
  const float* emb    = (const float*)d_in[2];
  const float* W_ih   = (const float*)d_in[3];
  const float* W_hh   = (const float*)d_in[4];
  const float* b_ih   = (const float*)d_in[5];
  const float* b_hh   = (const float*)d_in[6];
  const float* W_lin  = (const float*)d_in[7];
  const float* b_lin  = (const float*)d_in[8];
  float* out = (float*)d_out;

  char* ws = (char*)d_ws;
  size_t off = 256;                                   // barrier counter first
  const size_t WIH_OFF = off; off += (size_t)G4H * E_ * 2;          // 4 MB
  const size_t WHH_OFF = off; off += (size_t)G4H * H_ * 2;          // 8 MB
  const size_t WO_OFF  = off; off += (size_t)O_ * H_ * 2;           // 256 KB
  const size_t XP_OFF  = off; off += (size_t)B_ * T_ * G4H * 2;     // 128 MB
  const size_t HSP_OFF = off; off += (size_t)B_ * T_ * H_ * 2;      // 32 MB
  const size_t HP_OFF  = off; off += (size_t)2 * 2 * 32 * TILE_ELEMS * 2; // 128K
  const size_t VP_OFF  = off;                                       // 16 KB

  unsigned int* bar = (unsigned int*)(ws);
  __bf16* wih_pack  = (__bf16*)(ws + WIH_OFF);
  __bf16* whh_pack  = (__bf16*)(ws + WHH_OFF);
  __bf16* wo_pack   = (__bf16*)(ws + WO_OFF);
  __bf16* xproj2    = (__bf16*)(ws + XP_OFF);
  __bf16* hs_pack   = (__bf16*)(ws + HSP_OFF);
  __bf16* hpack     = (__bf16*)(ws + HP_OFF);
  float*  vproj     = (float*)(ws + VP_OFF);

  hipMemsetAsync(ws, 0, 256, stream);   // grid-barrier counter

  pack_b_kernel<<<(G4H * E_) / 256, 256, 0, stream>>>(W_ih, wih_pack,
                                                      E_ / 32, E_, 0);
  pack_b_kernel<<<(G4H * H_) / 256, 256, 0, stream>>>(W_hh, whh_pack,
                                                      H_ / 32, H_, 0);
  pack_b_kernel<<<(O_ * H_) / 256, 256, 0, stream>>>(W_lin, wo_pack,
                                                     H_ / 32, 2 * H_, H_);

  embed_xproj_kernel<<<dim3((B_ * T_) / 16, G4H / 128), 256, 0, stream>>>(
      tokens, emb, wih_pack, b_ih, b_hh, xproj2);
  lstm_persistent_kernel<<<NWG_LSTM, 256, 0, stream>>>(whh_pack, xproj2, hpack,
                                                       hs_pack, bar);
  vproj_kernel<<<(B_ * O_) / 256, 256, 0, stream>>>(vidx, hs_pack, W_lin,
                                                    b_lin, vproj);
  logits_kernel<<<(B_ * T_) / 16, 256, 0, stream>>>(hs_pack, wo_pack, vproj,
                                                    out);
}
